// TransformerEncoderBlock_23596550324701
// MI455X (gfx1250) — matrix operand order repaired and checked by reading, not compiled, not run on hardware
//
#include <hip/hip_runtime.h>
#include <hip/hip_bf16.h>

typedef __bf16 bf16;
typedef __attribute__((ext_vector_type(16))) __bf16 bf16x16;
typedef __attribute__((ext_vector_type(8)))  __bf16 bf16x8;
typedef __attribute__((ext_vector_type(8)))  float  f32x8;
typedef __attribute__((ext_vector_type(4)))  float  f32x4;

#define CDIM   512
#define TDIM   4096
#define NBATCH 8
#define CH     64
#define NHEAD  8
#define DH     64
#define NCHUNK (TDIM / CH)
#define KSTEPS (CDIM / 32)      // 16 K-steps of 32 over C
#define SCALE  0.125f           // 1/sqrt(64)
#define LDW    72               // padded bf16 row stride (144 B = 36 dwords: conflict-free)
#define SCW    68               // padded f32 row stride (272 B = 68 dwords: conflict-free)

static __device__ inline bf16x16 join8(bf16x8 lo, bf16x8 hi) {
  bf16x16 r;
#pragma unroll
  for (int i = 0; i < 8; ++i) { r[i] = lo[i]; r[i + 8] = hi[i]; }
  return r;
}

static __device__ inline f32x8 wmma_bf16(bf16x16 a, bf16x16 b, f32x8 c) {
  return __builtin_amdgcn_wmma_f32_16x16x32_bf16(
      /*neg_a=*/false, a, /*neg_b=*/false, b,
      /*c_mod=*/(short)0, c, /*reuse_a=*/false, /*reuse_b=*/false);
}

// ---------------------------------------------------------------------------
// Weight prep:
//  mats 0..3 (Wq,Wk,Wv,Wo): bf16 in WMMA A-fragment order.
//    Fragment (dtile, ktile): lane l (m = l&15, half = l>>4), element e:
//      d = dtile*16 + m ;  k = ktile*32 + e + 16*half
//    stored frag*512 + lane*16 + e.
//  mat 4: plain row-major bf16 copy of Wv (B-operand for transposed V proj).
// ---------------------------------------------------------------------------
__global__ void prep_w(const float* __restrict__ Wq, const float* __restrict__ Wk,
                       const float* __restrict__ Wv, const float* __restrict__ Wo,
                       bf16* __restrict__ wsw) {
  int idx = blockIdx.x * 256 + threadIdx.x;   // 0 .. 5*512*512-1
  int mat = idx >> 18;
  int rem = idx & 262143;
  if (mat == 4) { wsw[idx] = (bf16)Wv[rem]; return; }
  int frag   = rem >> 9;
  int within = rem & 511;
  int lane = within >> 4, e = within & 15;
  int dtile = frag >> 4, ktile = frag & 15;
  int m = lane & 15, half = lane >> 4;
  int d = dtile * 16 + m;
  int k = ktile * 32 + e + 16 * half;
  const float* W = (mat == 0) ? Wq : (mat == 1) ? Wk : (mat == 2) ? Wv : Wo;
  wsw[idx] = (bf16)W[d * CDIM + k];
}

// x [N, C, T] fp32 -> xT [(n*T + t), C] bf16
__global__ void prep_x(const float* __restrict__ x, bf16* __restrict__ xT) {
  int idx = blockIdx.x * 256 + threadIdx.x;
  int n = idx / (CDIM * TDIM);
  int rem = idx - n * (CDIM * TDIM);
  int c = rem / TDIM, t = rem - c * TDIM;
  xT[(size_t)(n * TDIM + t) * CDIM + c] = (bf16)x[idx];
}

// ---------------------------------------------------------------------------
// Fused block-local MHA + residual. One WG (16 wave32, 512 thr) per (n, chunk).
// ---------------------------------------------------------------------------
__global__ __launch_bounds__(512, 1) void attn_kernel(
    const float* __restrict__ x, const float* __restrict__ masks,
    const float* __restrict__ bq, const float* __restrict__ bk,
    const float* __restrict__ bv, const float* __restrict__ bo,
    const bf16* __restrict__ wsw, const bf16* __restrict__ xT,
    float* __restrict__ out) {
  __shared__ __align__(16) bf16  qs [CH * LDW];  // [q_t][dh]
  __shared__ __align__(16) bf16  kts[CH * LDW];  // [k_t][dh]
  __shared__ __align__(16) bf16  vts[DH * LDW];  // [dh][k_t]
  __shared__ __align__(16) float sc [CH * SCW];  // [q_t][k_t] scores fp32
  __shared__ __align__(16) bf16  ps [CH * LDW];  // [q_t][k_t] probs
  __shared__ __align__(16) bf16  ah [CH * LDW];  // [q_t][dh]  per-head attn out

  const int tid  = threadIdx.x;
  const int wave = tid >> 5;                     // 0..15
  const int lane = tid & 31;
  const int l15  = lane & 15;
  const int half = lane >> 4;
  const int n     = blockIdx.x >> 6;
  const int chunk = blockIdx.x & 63;
  const int t0    = chunk * CH;
  const int mtile = wave >> 2, ntile = wave & 3; // per-wave 16x16 tile coords

  const bf16* Wq_s = wsw;
  const bf16* Wk_s = wsw + 1 * CDIM * CDIM;
  const bf16* Wo_s = wsw + 3 * CDIM * CDIM;
  const bf16* Wv_p = wsw + 4 * CDIM * CDIM;      // plain row-major bf16 Wv
  const bf16* xTn  = xT + (size_t)(n * TDIM + t0) * CDIM;

  // O accumulators: wave owns d' in [wave*32, wave*32+32): 2 dtiles x 4 ntiles.
  f32x8 accO[2][4];
#pragma unroll
  for (int i = 0; i < 2; ++i)
#pragma unroll
    for (int j = 0; j < 4; ++j) accO[i][j] = (f32x8){};

  for (int h = 0; h < NHEAD; ++h) {
    // prefetch this head's Wo fragments used by P4 (global_prefetch_b8)
#pragma unroll
    for (int i = 0; i < 2; ++i)
      __builtin_prefetch(Wo_s + (((wave * 2 + i) * 16 + h * 2) * 512 + lane * 16), 0, 0);

    // ---- P1a: Q and K head tiles (M = dh, N = t), shared B fragments ----
    {
      int dtileG = h * 4 + mtile;
      f32x8 accq = (f32x8){}, acck = (f32x8){};
#pragma unroll 4
      for (int kt = 0; kt < KSTEPS; ++kt) {
        bf16x16 b  = *(const bf16x16*)(xTn + (size_t)(ntile * 16 + l15) * CDIM +
                                       kt * 32 + half * 16);
        bf16x16 aq = *(const bf16x16*)(Wq_s + ((dtileG * 16 + kt) * 512 + lane * 16));
        bf16x16 ak = *(const bf16x16*)(Wk_s + ((dtileG * 16 + kt) * 512 + lane * 16));
        accq = wmma_bf16(aq, b, accq);
        acck = wmma_bf16(ak, b, acck);
      }
      bf16x8 pq, pk;
#pragma unroll
      for (int r = 0; r < 8; ++r) {
        int dhi = mtile * 16 + 8 * half + r;
        pq[r] = (bf16)(accq[r] + bq[h * DH + dhi]);
        pk[r] = (bf16)(acck[r] + bk[h * DH + dhi]);
      }
      int rowoff = (ntile * 16 + l15) * LDW + mtile * 16 + 8 * half;
      *(bf16x8*)(qs  + rowoff) = pq;
      *(bf16x8*)(kts + rowoff) = pk;
    }
    // ---- P1b: V head tile transposed (M = t, N = dh): A = x^T, B = Wv rows ----
    {
      int dhl = ntile * 16 + l15;                // dh within head
      int dhg = h * DH + dhl;                    // global row of Wv
      f32x8 acc = (f32x8){};
#pragma unroll 4
      for (int kt = 0; kt < KSTEPS; ++kt) {
        const bf16* ar = xTn + (size_t)(mtile * 16 + l15) * CDIM + kt * 32 + half * 16;
        bf16x16 a = join8(*(const bf16x8*)ar, *(const bf16x8*)(ar + 8));
        bf16x16 b = *(const bf16x16*)(Wv_p + (size_t)dhg * CDIM + kt * 32 + half * 16);
        acc = wmma_bf16(a, b, acc);
      }
      float bvv = bv[dhg];
      bf16x8 pk;
#pragma unroll
      for (int r = 0; r < 8; ++r) pk[r] = (bf16)(acc[r] + bvv);
      *(bf16x8*)(vts + dhl * LDW + mtile * 16 + 8 * half) = pk;
    }
    __syncthreads();

    // ---- P2: scores^T = K·Q^T (M = k_t, N = q_t), contiguous f32 stores ----
    {
      f32x8 acc = (f32x8){};
#pragma unroll
      for (int kt = 0; kt < 2; ++kt) {
        int k32 = kt * 32;
        const bf16* ar = kts + (mtile * 16 + l15) * LDW + k32 + half * 16;
        bf16x16 a = join8(*(const bf16x8*)ar, *(const bf16x8*)(ar + 8));
        bf16x16 b = *(const bf16x16*)(qs + (ntile * 16 + l15) * LDW + k32 + half * 16);
        acc = wmma_bf16(a, b, acc);
      }
      const float* mrow = masks + n * TDIM + t0 + mtile * 16 + 8 * half;
      f32x4 m0 = *(const f32x4*)mrow;
      f32x4 m1 = *(const f32x4*)(mrow + 4);
      f32x4 s0, s1;
#pragma unroll
      for (int r = 0; r < 4; ++r) {
        s0[r] = (m0[r] > 0.f) ? acc[r] * SCALE : -1e9f;
        s1[r] = (m1[r] > 0.f) ? acc[r + 4] * SCALE : -1e9f;
      }
      float* dst = sc + (ntile * 16 + l15) * SCW + mtile * 16 + 8 * half;
      ((f32x4*)dst)[0] = s0;
      ((f32x4*)dst)[1] = s1;
    }
    __syncthreads();

    // ---- softmax over key dim, one row per thread ----
    if (tid < CH) {
      float* row = sc + tid * SCW;
      float mx = -3.4e38f;
      for (int k2 = 0; k2 < CH; ++k2) mx = fmaxf(mx, row[k2]);
      float s = 0.f;
      for (int k2 = 0; k2 < CH; ++k2) { float e = __expf(row[k2] - mx); row[k2] = e; s += e; }
      float inv = 1.f / s;
      for (int k2 = 0; k2 < CH; ++k2) ps[tid * LDW + k2] = (bf16)(row[k2] * inv);
    }
    __syncthreads();

    // ---- P3: attn_h^T = V^T·P^T (M = dh, N = q_t), packed bf16 stores ----
    {
      f32x8 acc = (f32x8){};
#pragma unroll
      for (int kt = 0; kt < 2; ++kt) {
        int k32 = kt * 32;
        const bf16* ar = vts + (mtile * 16 + l15) * LDW + k32 + half * 16;
        bf16x16 a = join8(*(const bf16x8*)ar, *(const bf16x8*)(ar + 8));
        bf16x16 b = *(const bf16x16*)(ps + (ntile * 16 + l15) * LDW + k32 + half * 16);
        acc = wmma_bf16(a, b, acc);
      }
      bf16x8 pk;
#pragma unroll
      for (int r = 0; r < 8; ++r) pk[r] = (bf16)acc[r];
      *(bf16x8*)(ah + (ntile * 16 + l15) * LDW + mtile * 16 + 8 * half) = pk;
    }
    __syncthreads();

    // ---- P4 (interleaved): accumulate this head into O registers ----
#pragma unroll
    for (int i = 0; i < 2; ++i) {
      int dtileG = wave * 2 + i;                 // d' tile 0..31
#pragma unroll
      for (int j = 0; j < 4; ++j) {
#pragma unroll
        for (int kt2 = 0; kt2 < 2; ++kt2) {
          int ktG = h * 2 + kt2;
          bf16x16 a = *(const bf16x16*)(Wo_s + ((dtileG * 16 + ktG) * 512 + lane * 16));
          bf16x16 b = *(const bf16x16*)(ah + (j * 16 + l15) * LDW + kt2 * 32 + half * 16);
          accO[i][j] = wmma_bf16(a, b, accO[i][j]);
        }
      }
    }
    // next-head LDS reuse is fenced by the barriers above
  }

  // ---- epilogue: bias, mask, residual, store ----
#pragma unroll
  for (int i = 0; i < 2; ++i) {
    int dtileG = wave * 2 + i;
#pragma unroll
    for (int j = 0; j < 4; ++j) {
      int tcol = t0 + j * 16 + l15;
      float mval = masks[n * TDIM + tcol];
#pragma unroll
      for (int r = 0; r < 8; ++r) {
        int d = dtileG * 16 + r + 8 * half;
        size_t g = (size_t)n * CDIM * TDIM + (size_t)d * TDIM + tcol;
        out[g] = (accO[i][j][r] + bo[d]) * mval + x[g];
      }
    }
  }
}

// ---------------------------------------------------------------------------
// InstanceNorm1d over T per (n, c) row, in place on d_out.
// ---------------------------------------------------------------------------
__global__ void norm_kernel(float* __restrict__ out) {
  __shared__ float r1[256], r2[256];
  int row = blockIdx.x;
  float* p = out + (size_t)row * TDIM;
  int tid = threadIdx.x;
  float s1 = 0.f, s2 = 0.f;
  for (int i = tid; i < TDIM; i += 256) { float v = p[i]; s1 += v; s2 += v * v; }
  r1[tid] = s1; r2[tid] = s2;
  __syncthreads();
  for (int off = 128; off > 0; off >>= 1) {
    if (tid < off) { r1[tid] += r1[tid + off]; r2[tid] += r2[tid + off]; }
    __syncthreads();
  }
  float mean = r1[0] * (1.f / TDIM);
  float var  = r2[0] * (1.f / TDIM) - mean * mean;
  float inv  = rsqrtf(var + 1e-5f);
  for (int i = tid; i < TDIM; i += 256) p[i] = (p[i] - mean) * inv;
}

extern "C" void kernel_launch(void* const* d_in, const int* in_sizes, int n_in,
                              void* d_out, int out_size, void* d_ws, size_t ws_size,
                              hipStream_t stream) {
  (void)in_sizes; (void)n_in; (void)out_size; (void)ws_size;
  const float* x  = (const float*)d_in[0];
  const float* mk = (const float*)d_in[1];
  const float* Wq = (const float*)d_in[2];
  const float* bq = (const float*)d_in[3];
  const float* Wk = (const float*)d_in[4];
  const float* bk = (const float*)d_in[5];
  const float* Wv = (const float*)d_in[6];
  const float* bv = (const float*)d_in[7];
  const float* Wo = (const float*)d_in[8];
  const float* bo = (const float*)d_in[9];
  float* out = (float*)d_out;

  bf16* wsw = (bf16*)d_ws;                       // 5*C*C bf16 = 2.5 MB
  bf16* xT  = wsw + 5 * CDIM * CDIM;             // N*T*C bf16 = 32 MB

  prep_w<<<(5 * CDIM * CDIM) / 256, 256, 0, stream>>>(Wq, Wk, Wv, Wo, wsw);
  prep_x<<<(NBATCH * CDIM * TDIM) / 256, 256, 0, stream>>>(x, xT);
  attn_kernel<<<NBATCH * NCHUNK, 512, 0, stream>>>(x, mk, bq, bk, bv, bo, wsw, xT, out);
  norm_kernel<<<NBATCH * CDIM, 256, 0, stream>>>(out);
}
